// Attention_25847113187663
// MI455X (gfx1250) — matrix operand order repaired and checked by reading, not compiled, not run on hardware
//
#include <hip/hip_runtime.h>
#include <hip/hip_bf16.h>
#include <stdint.h>

// Problem constants (from reference): B=2, N=2048, C=512, H=8, D=64
#define NB   2
#define NN   2048
#define NC   512
#define NH   8
#define ND   64
#define NM1  2047            // N-1 (bias sub-block)
#define QSCALE 0.125f        // D^-0.5
#define KBLK (NN / 32)       // 64 key blocks of 32

typedef __bf16 v16bf __attribute__((ext_vector_type(16)));
typedef __bf16 v2bf  __attribute__((ext_vector_type(2)));
typedef float  v8f   __attribute__((ext_vector_type(8)));

union FragU { v16bf v; unsigned u[8]; };

// ---------------- bf16 conversion (packed hw cvt) -------------------------
__device__ __forceinline__ unsigned pack2(float a, float b) {
#if __has_builtin(__builtin_amdgcn_cvt_pk_bf16_f32)
  union { v2bf v; unsigned u; } cv;
  cv.v = __builtin_amdgcn_cvt_pk_bf16_f32(a, b);
  return cv.u;
#else
  // builtin not exposed on this toolchain; the hw op exists on gfx1250
  unsigned r;
  asm("v_cvt_pk_bf16_f32 %0, %1, %2" : "=v"(r) : "v"(a), "v"(b));
  return r;
#endif
}
__device__ __forceinline__ unsigned short f32_to_bf16(float f) {
  return (unsigned short)(pack2(f, f) & 0xffffu);
}

// ---------------- async global->LDS (CDNA5 ASYNCcnt path) -----------------
#if __has_builtin(__builtin_amdgcn_global_load_async_to_lds_b128)
#define HAVE_ASYNC 1
// exact pointee type from the builtin's signature: vector_size(16) int
typedef int b128_t __attribute__((vector_size(4 * sizeof(int))));
typedef __attribute__((address_space(1))) b128_t g_b128;
typedef __attribute__((address_space(3))) b128_t l_b128;
__device__ __forceinline__ void async_b128(void* lds, const void* g) {
  // generic->AS casts via integer: AS(3) pointers are the 32-bit LDS offset
  __builtin_amdgcn_global_load_async_to_lds_b128(
      (g_b128*)(uintptr_t)g, (l_b128*)(unsigned)(uintptr_t)lds, 0, 0);
}
__device__ __forceinline__ void wait_async0() {
#if __has_builtin(__builtin_amdgcn_s_wait_asynccnt)
  __builtin_amdgcn_s_wait_asynccnt(0);
#else
  asm volatile("s_wait_asynccnt 0" ::: "memory");
#endif
}
#endif

__device__ __forceinline__ v8f wmma_bf16(v16bf a, v16bf b, v8f c) {
  // D = A(16x32 bf16) x B(32x16 bf16) + C(16x16 f32)
  return __builtin_amdgcn_wmma_f32_16x16x32_bf16(false, a, false, b, (short)0, c,
                                                 false, false);
}

// A-fragment 16x32 (bf16) from an M-major, K-contiguous tile.
// ISA layout: lanes 0-15 M=0-15 (K in {0..7,16..23}); lanes 16-31 same M
// (K in {8..15,24..31}); VGPR i holds K-pair 2i + 8*((i>>2)+half).
__device__ __forceinline__ v16bf load_a_frag(const unsigned short* src, int stride, int lane) {
  const int m = lane & 15, half = lane >> 4;
  FragU f;
#pragma unroll
  for (int i = 0; i < 8; ++i) {
    const int k = 2 * i + 8 * ((i >> 2) + half);
    f.u[i] = *(const unsigned*)(src + m * stride + k);
  }
  return f.v;
}

// B-fragment 32x16 (bf16) from an N-major, K-contiguous tile.
// Layout matches A: lanes 0-15 K in {0..7,16..23}, lanes 16-31 K in {8..15,24..31}; VGPR j holds K-pair 2j + 8*((j>>2)+half).
__device__ __forceinline__ v16bf load_b_frag(const unsigned short* src, int stride, int lane) {
  const int n = lane & 15, g = lane >> 4;
  FragU f;
#pragma unroll
  for (int j = 0; j < 8; ++j) {
    const int k = 2 * j + 8 * ((j >> 2) + g);
    f.u[j] = *(const unsigned*)(src + n * stride + k);
  }
  return f.v;
}

// ---------------------------------------------------------------------------
// Shared GEMM body: C[128x64] tile of  A[., 512] x W[., 512]^T  (K = 512).
// A and W are f32 in global; converted to bf16 while staging into LDS.
// ---------------------------------------------------------------------------
__device__ __forceinline__ void gemm_128x64_k512(const float* __restrict__ A,
                                                 const float* __restrict__ W,
                                                 unsigned short* lds_x,   // 128*32
                                                 unsigned short* lds_w,   // 64*32
                                                 int lane, int w, v8f c[4]) {
#pragma unroll
  for (int f = 0; f < 4; ++f) c[f] = (v8f){0.f, 0.f, 0.f, 0.f, 0.f, 0.f, 0.f, 0.f};

  const int tid = threadIdx.x;
  for (int kk = 0; kk < NC; kk += 32) {
    __syncthreads();
    // stage A tile: 128 rows x 32 cols -> 2048 bf16-pairs
#pragma unroll
    for (int it = 0; it < 8; ++it) {
      const int i = tid + it * 256;
      const int row = i >> 4, cp = i & 15;
      const float2 v = *(const float2*)(A + (size_t)row * NC + kk + 2 * cp);
      ((unsigned*)lds_x)[i] = pack2(v.x, v.y);
    }
    // stage W tile: 64 rows x 32 cols -> 1024 bf16-pairs
#pragma unroll
    for (int it = 0; it < 4; ++it) {
      const int i = tid + it * 256;
      const int row = i >> 4, cp = i & 15;
      const float2 v = *(const float2*)(W + (size_t)row * NC + kk + 2 * cp);
      ((unsigned*)lds_w)[i] = pack2(v.x, v.y);
    }
    __syncthreads();

    const v16bf a  = load_a_frag(lds_x + (w * 16) * 32, 32, lane);
    const v16bf b0 = load_b_frag(lds_w + 0 * 16 * 32, 32, lane);
    const v16bf b1 = load_b_frag(lds_w + 1 * 16 * 32, 32, lane);
    const v16bf b2 = load_b_frag(lds_w + 2 * 16 * 32, 32, lane);
    const v16bf b3 = load_b_frag(lds_w + 3 * 16 * 32, 32, lane);
    c[0] = wmma_bf16(a, b0, c[0]);
    c[1] = wmma_bf16(a, b1, c[1]);
    c[2] = wmma_bf16(a, b2, c[2]);
    c[3] = wmma_bf16(a, b3, c[3]);
  }
}

// ---------------------------------------------------------------------------
// Kernel 1: QKV projection. qkv = x @ qkv_w.T, scattered to q/k/v [B,H,N,D]
// as bf16; Q pre-scaled by D^-0.5.  grid (32, 24), block 256.
// ---------------------------------------------------------------------------
__global__ __launch_bounds__(256) void qkv_gemm_kernel(
    const float* __restrict__ x, const float* __restrict__ qkv_w,
    unsigned short* __restrict__ qb, unsigned short* __restrict__ kb,
    unsigned short* __restrict__ vb) {
  __shared__ __align__(16) unsigned short lds_x[128 * 32];
  __shared__ __align__(16) unsigned short lds_w[64 * 32];
  const int lane = threadIdx.x & 31, w = threadIdx.x >> 5, half = (threadIdx.x >> 4) & 1;
  const int mbase = blockIdx.x * 128, nbase = blockIdx.y * 64;

  v8f c[4];
  gemm_128x64_k512(x + (size_t)mbase * NC, qkv_w + (size_t)nbase * NC, lds_x, lds_w,
                   lane, w, c);

#pragma unroll
  for (int f = 0; f < 4; ++f) {
    const int n = nbase + (lane & 15) + f * 16;      // 0..1535
    const int which = n >> 9;                         // 0=q 1=k 2=v
    const int hh = (n >> 6) & 7, d = n & 63;
    unsigned short* dst = (which == 0) ? qb : ((which == 1) ? kb : vb);
    const float sc = (which == 0) ? QSCALE : 1.0f;
#pragma unroll
    for (int r = 0; r < 8; ++r) {
      const int m = mbase + w * 16 + r + 8 * half;    // 0..4095
      const int bb = m >> 11, row = m & (NN - 1);
      dst[(((size_t)bb * NH + hh) * NN + row) * ND + d] = f32_to_bf16(c[f][r] * sc);
    }
  }
}

// ---------------------------------------------------------------------------
// K/V tile staging for one 32-key block: K flat (async when available),
// V transposed through VGPRs -> lds_vt[d][key].
// ---------------------------------------------------------------------------
__device__ __forceinline__ void stage_kv(const unsigned short* kg,
                                         const unsigned short* vg,
                                         unsigned short* dst_k,
                                         unsigned short* dst_vt,
                                         int kb, int tid) {
  const unsigned short* ksrc = kg + (size_t)kb * 32 * ND;
#ifdef HAVE_ASYNC
  async_b128((char*)dst_k + tid * 16, (const char*)ksrc + tid * 16);
#else
  ((uint4*)dst_k)[tid] = ((const uint4*)ksrc)[tid];
#endif
  const unsigned* vsrc = (const unsigned*)(vg + (size_t)kb * 32 * ND);
#pragma unroll
  for (int it = 0; it < 4; ++it) {
    const int i = tid + it * 256;            // bf16-pair index
    const int key = i >> 5, d = (i & 31) * 2;
    const unsigned vv = vsrc[i];
    dst_vt[d * 32 + key] = (unsigned short)(vv & 0xffffu);
    dst_vt[(d + 1) * 32 + key] = (unsigned short)(vv >> 16);
  }
}

// ---------------------------------------------------------------------------
// Kernel 2: fused flash attention with additive bias.
// grid (N/128, B*H) = (16, 16), block 256 (8 waves x 16 query rows).
// Double-buffered K/V tiles; async global->LDS overlap with WMMA compute.
// ---------------------------------------------------------------------------
__global__ __launch_bounds__(256) void flash_attn_kernel(
    const unsigned short* __restrict__ qb, const unsigned short* __restrict__ kbuf,
    const unsigned short* __restrict__ vbuf, const float* __restrict__ adj,
    float* __restrict__ attn_out) {
  __shared__ __align__(16) unsigned short lds_q[128 * 64];      // 16 KB
  __shared__ __align__(16) unsigned short lds_k[2][32 * 64];    //  8 KB  [key][d]
  __shared__ __align__(16) unsigned short lds_vt[2][64 * 32];   //  8 KB  [d][key]
  __shared__ __align__(16) unsigned short lds_p[8][16 * 32];    //  8 KB  per-wave P

  const int tid = threadIdx.x;
  const int lane = tid & 31, w = tid >> 5, half = (lane >> 4);
  const int bh = blockIdx.y;
  const int b = bh >> 3, h = bh & 7;
  const int row0 = blockIdx.x * 128;

  const unsigned short* qg = qb + ((size_t)bh * NN + row0) * ND;
  const unsigned short* kg = kbuf + (size_t)bh * NN * ND;
  const unsigned short* vg = vbuf + (size_t)bh * NN * ND;
  const float* adjb = adj + (size_t)b * NM1 * NM1;

  // stage Q block (128x64 bf16) + first K/V tile
#ifdef HAVE_ASYNC
#pragma unroll
  for (int it = 0; it < 4; ++it) {
    const int i = tid + it * 256;
    async_b128((char*)lds_q + i * 16, (const char*)qg + i * 16);
  }
#else
  {
    const uint4* src = (const uint4*)qg;
    uint4* dst = (uint4*)lds_q;
#pragma unroll
    for (int it = 0; it < 4; ++it) dst[tid + it * 256] = src[tid + it * 256];
  }
#endif
  stage_kv(kg, vg, lds_k[0], lds_vt[0], 0, tid);
#ifdef HAVE_ASYNC
  wait_async0();
#endif
  __syncthreads();

  const v16bf qa0 = load_a_frag(lds_q + (w * 16) * ND + 0, ND, lane);
  const v16bf qa1 = load_a_frag(lds_q + (w * 16) * ND + 32, ND, lane);

  float m_run[8], l_run[8];
  v8f o[4];
#pragma unroll
  for (int r = 0; r < 8; ++r) { m_run[r] = -3.0e38f; l_run[r] = 0.f; }
#pragma unroll
  for (int f = 0; f < 4; ++f) o[f] = (v8f){0.f, 0.f, 0.f, 0.f, 0.f, 0.f, 0.f, 0.f};

  for (int kb = 0; kb < KBLK; ++kb) {
    const int cur = kb & 1;
    // prefetch/stage next tile while computing on current one
    if (kb + 1 < KBLK) stage_kv(kg, vg, lds_k[cur ^ 1], lds_vt[cur ^ 1], kb + 1, tid);

    // S = Q K^T for 16 rows x 32 keys (f32); scale folded into Q
    const unsigned short* kc = lds_k[cur];
    const v16bf kb00 = load_b_frag(kc + 0, ND, lane);            // keys 0-15, d 0-31
    const v16bf kb01 = load_b_frag(kc + 32, ND, lane);           // keys 0-15, d 32-63
    const v16bf kb10 = load_b_frag(kc + 16 * ND + 0, ND, lane);  // keys 16-31
    const v16bf kb11 = load_b_frag(kc + 16 * ND + 32, ND, lane);
    v8f s0 = (v8f){0.f, 0.f, 0.f, 0.f, 0.f, 0.f, 0.f, 0.f};
    v8f s1 = s0;
    s0 = wmma_bf16(qa0, kb00, s0);
    s0 = wmma_bf16(qa1, kb01, s0);
    s1 = wmma_bf16(qa0, kb10, s1);
    s1 = wmma_bf16(qa1, kb11, s1);

    // additive positional bias (f32, zero on padded last row/col)
    const int j0 = kb * 32 + (lane & 15);
#pragma unroll
    for (int r = 0; r < 8; ++r) {
      const int i = row0 + w * 16 + r + 8 * half;
      if (i < NM1) {
        if (j0 < NM1) s0[r] += adjb[(size_t)i * NM1 + j0];
        if (j0 + 16 < NM1) s1[r] += adjb[(size_t)i * NM1 + j0 + 16];
      }
    }
    // prefetch next bias rows into cache (global_prefetch_b8)
    if (kb + 1 < KBLK) {
      const int jn = (kb + 1) * 32 + (lane & 15);
#pragma unroll
      for (int r = 0; r < 8; ++r) {
        const int i = row0 + w * 16 + r + 8 * half;
        if (i < NM1 && jn < NM1) __builtin_prefetch(adjb + (size_t)i * NM1 + jn, 0, 0);
      }
    }

    // online softmax update (row reductions across 16 lanes of each half)
    float alpha[8];
#pragma unroll
    for (int r = 0; r < 8; ++r) {
      float mx = fmaxf(s0[r], s1[r]);
      mx = fmaxf(mx, __shfl_xor(mx, 1, 32));
      mx = fmaxf(mx, __shfl_xor(mx, 2, 32));
      mx = fmaxf(mx, __shfl_xor(mx, 4, 32));
      mx = fmaxf(mx, __shfl_xor(mx, 8, 32));
      const float mnew = fmaxf(m_run[r], mx);
      alpha[r] = __expf(m_run[r] - mnew);
      m_run[r] = mnew;
    }
    unsigned short* pw = lds_p[w];
#pragma unroll
    for (int r = 0; r < 8; ++r) {
      const float p0 = __expf(s0[r] - m_run[r]);
      const float p1 = __expf(s1[r] - m_run[r]);
      float rs = p0 + p1;
      rs += __shfl_xor(rs, 1, 32);
      rs += __shfl_xor(rs, 2, 32);
      rs += __shfl_xor(rs, 4, 32);
      rs += __shfl_xor(rs, 8, 32);
      l_run[r] = l_run[r] * alpha[r] + rs;
      const unsigned pk = pack2(p0, p1);
      const int rr = r + 8 * half;
      pw[rr * 32 + (lane & 15)] = (unsigned short)(pk & 0xffffu);
      pw[rr * 32 + (lane & 15) + 16] = (unsigned short)(pk >> 16);
    }
#pragma unroll
    for (int f = 0; f < 4; ++f)
#pragma unroll
      for (int r = 0; r < 8; ++r) o[f][r] *= alpha[r];

    // O += P V  (P re-read from LDS in A-fragment layout; V^T is K-contiguous)
    const v16bf pa = load_a_frag(pw, 32, lane);
    const unsigned short* vt = lds_vt[cur];
    const v16bf vb0 = load_b_frag(vt + 0 * 16 * 32, 32, lane);
    const v16bf vb1 = load_b_frag(vt + 1 * 16 * 32, 32, lane);
    const v16bf vb2 = load_b_frag(vt + 2 * 16 * 32, 32, lane);
    const v16bf vb3 = load_b_frag(vt + 3 * 16 * 32, 32, lane);
    o[0] = wmma_bf16(pa, vb0, o[0]);
    o[1] = wmma_bf16(pa, vb1, o[1]);
    o[2] = wmma_bf16(pa, vb2, o[2]);
    o[3] = wmma_bf16(pa, vb3, o[3]);

    // make next tile visible to all waves before the next iteration
#ifdef HAVE_ASYNC
    wait_async0();
#endif
    __syncthreads();
  }

  // normalize and write [B,N,C] f32 (c = h*64 + d)
  float inv[8];
#pragma unroll
  for (int r = 0; r < 8; ++r) inv[r] = __builtin_amdgcn_rcpf(l_run[r]);
  float* outp = attn_out + ((size_t)b * NN + row0 + w * 16) * NC + h * ND;
#pragma unroll
  for (int f = 0; f < 4; ++f) {
#pragma unroll
    for (int r = 0; r < 8; ++r) {
      const int rr = r + 8 * half;
      outp[(size_t)rr * NC + (lane & 15) + f * 16] = o[f][r] * inv[r];
    }
  }
}

// ---------------------------------------------------------------------------
// Kernel 3: output projection. out = attn @ proj_w.T + proj_b  (f32 out).
// grid (32, 8), block 256.
// ---------------------------------------------------------------------------
__global__ __launch_bounds__(256) void proj_gemm_kernel(
    const float* __restrict__ ao, const float* __restrict__ proj_w,
    const float* __restrict__ proj_b, float* __restrict__ out) {
  __shared__ __align__(16) unsigned short lds_x[128 * 32];
  __shared__ __align__(16) unsigned short lds_w[64 * 32];
  const int lane = threadIdx.x & 31, w = threadIdx.x >> 5, half = (threadIdx.x >> 4) & 1;
  const int mbase = blockIdx.x * 128, nbase = blockIdx.y * 64;

  v8f c[4];
  gemm_128x64_k512(ao + (size_t)mbase * NC, proj_w + (size_t)nbase * NC, lds_x, lds_w,
                   lane, w, c);

#pragma unroll
  for (int f = 0; f < 4; ++f) {
    const int n = nbase + (lane & 15) + f * 16;
    const float bias = proj_b[n];
#pragma unroll
    for (int r = 0; r < 8; ++r) {
      const int m = mbase + w * 16 + r + 8 * half;
      out[(size_t)m * NC + n] = c[f][r] + bias;
    }
  }
}

// ---------------------------------------------------------------------------
// launch: inputs = { x, adj_pos_embed, qkv_w, proj_w, proj_b }
// workspace: q|k|v bf16 (4 MB each) + attn f32 (8 MB) = 20 MB
// ---------------------------------------------------------------------------
extern "C" void kernel_launch(void* const* d_in, const int* in_sizes, int n_in,
                              void* d_out, int out_size, void* d_ws, size_t ws_size,
                              hipStream_t stream) {
  (void)in_sizes; (void)n_in; (void)out_size; (void)ws_size;
  const float* x      = (const float*)d_in[0];
  const float* adj    = (const float*)d_in[1];
  const float* qkv_w  = (const float*)d_in[2];
  const float* proj_w = (const float*)d_in[3];
  const float* proj_b = (const float*)d_in[4];
  float* out = (float*)d_out;

  char* ws = (char*)d_ws;
  unsigned short* qb = (unsigned short*)(ws);
  unsigned short* kb = (unsigned short*)(ws + (size_t)4 * 1024 * 1024);
  unsigned short* vb = (unsigned short*)(ws + (size_t)8 * 1024 * 1024);
  float* ao          = (float*)(ws + (size_t)12 * 1024 * 1024);

  qkv_gemm_kernel<<<dim3(32, 24), 256, 0, stream>>>(x, qkv_w, qb, kb, vb);
  flash_attn_kernel<<<dim3(16, 16), 256, 0, stream>>>(qb, kb, vb, adj, ao);
  proj_gemm_kernel<<<dim3(32, 8), 256, 0, stream>>>(ao, proj_w, proj_b, out);
}
